// MoEExperts_84817014161794
// MI455X (gfx1250) — compile-verified, matrix operand order checked by reading, not run on hardware
//
#include <hip/hip_runtime.h>
#include <hip/hip_bf16.h>

// ---------------------------------------------------------------------------
// MoE SwiGLU experts for MI455X (gfx1250, wave32, WMMA).
// Roofline: 206 GFLOP useful work vs ~470MB traffic @23.3TB/s -> matrix-core
// bound. Strategy: route tokens into per-expert padded segments, then run
// dense bf16 WMMA GEMM tiles (v_wmma_f32_16x16x32_bf16, fp32 accum).
// FFN2's A-tile (bf16 h buffer) streams through the gfx1250 async
// global->LDS path (GLOBAL_LOAD_ASYNC_TO_LDS_B128 + s_wait_asynccnt).
// ---------------------------------------------------------------------------

typedef __bf16 bf16;
typedef __attribute__((ext_vector_type(16))) __bf16        v16bf;
typedef __attribute__((ext_vector_type(8)))  float         v8f;
typedef __attribute__((ext_vector_type(4)))  float         f32x4;
typedef __attribute__((ext_vector_type(4)))  unsigned int  u32x4;

#define NE 8       // experts
#define DD 1024    // model dim
#define FF 4096    // ffn dim
#define TM 128     // tile M (tokens)
#define TN 128     // tile N
#define TK 32      // tile K (one bf16 WMMA K-step)

union Frag  { u32x4 q[2]; v16bf v; };   // 32B WMMA operand (16 x bf16)
union Pack8 { u32x4 q; bf16 h[8]; };    // 8 bf16 packed for b128 LDS store

// ---------------- gfx1250 async global->LDS helpers -------------------------

// One 16B lane-copy: VDST = 32-bit LDS byte address, VADDR = 64-bit global.
__device__ inline void async_copy_b128(const void* gptr, void* lptr) {
  unsigned lds_addr = (unsigned)(size_t)lptr;  // flat LDS addr truncates to offset
  asm volatile("global_load_async_to_lds_b128 %0, %1, off"
               :: "v"(lds_addr), "v"(gptr)
               : "memory");
}

__device__ inline void async_wait_all() {
  asm volatile("s_wait_asynccnt 0x0" ::: "memory");
}

// ---------------- routing pipeline -----------------------------------------

__global__ void moe_init(int* __restrict__ token_list, int* __restrict__ counts,
                         int Tpad) {
  int i = blockIdx.x * blockDim.x + threadIdx.x;
  if (i < Tpad) token_list[i] = -1;
  if (i < NE) counts[i] = 0;
}

__global__ void moe_count(const int* __restrict__ idx, int* __restrict__ counts,
                          int T) {
  int t = blockIdx.x * blockDim.x + threadIdx.x;
  if (t < T) atomicAdd(&counts[idx[t] & (NE - 1)], 1);
}

__global__ void moe_scan(const int* __restrict__ counts, int* __restrict__ offP,
                         int* __restrict__ cursor) {
  if (threadIdx.x == 0 && blockIdx.x == 0) {
    int o = 0;
    for (int e = 0; e < NE; ++e) {
      offP[e] = o;
      cursor[e] = o;
      o += ((counts[e] + TM - 1) / TM) * TM;  // pad segment to tile M
    }
    offP[NE] = o;
  }
}

__global__ void moe_scatter(const int* __restrict__ idx, int* __restrict__ cursor,
                            int* __restrict__ token_list, int T) {
  int t = blockIdx.x * blockDim.x + threadIdx.x;
  if (t < T) {
    int e = idx[t] & (NE - 1);
    int p = atomicAdd(&cursor[e], 1);
    token_list[p] = t;
  }
}

// ---------------- shared helpers --------------------------------------------

// Load a 16x32 bf16 operand fragment from a row-major [rows][TK] LDS tile.
// ISA layout: lanes 0-15 hold rows, K runs {kg*8..kg*8+7, 16+kg*8..} with
// kg = lane>>4 -> two 16-byte ds_load_b128 per lane.
__device__ inline v16bf load_frag(const bf16* lds_tile, int row, int kg) {
  const u32x4* p = (const u32x4*)(lds_tile + row * TK);
  Frag f;
  f.q[0] = p[kg];
  f.q[1] = p[2 + kg];
  return f.v;
}

__device__ inline int find_expert(const int* __restrict__ offP, int r0) {
  int e = -1;
#pragma unroll
  for (int i = 0; i < NE; ++i)
    if (r0 >= offP[i] && r0 < offP[i + 1]) e = i;
  return e;
}

// ---------------- FFN1: h = silu(x @ Wg) * (x @ Wu), bf16 out ---------------

__global__ __launch_bounds__(256)
void moe_ffn1(const float* __restrict__ x, const float* __restrict__ wg,
              const float* __restrict__ wu, const int* __restrict__ token_list,
              const int* __restrict__ offP, bf16* __restrict__ hbuf) {
  __shared__ __align__(16) bf16 lA[TM * TK];
  __shared__ __align__(16) bf16 lG[TN * TK];
  __shared__ __align__(16) bf16 lU[TN * TK];

  const int r0 = blockIdx.x * TM;   // compacted-token row base
  const int f0 = blockIdx.y * TN;   // ffn-column base
  const int e = find_expert(offP, r0);
  if (e < 0) return;                // past last padded segment (uniform)

  const int tid  = threadIdx.x;
  const int lane = tid & 31;
  const int wid  = tid >> 5;
  const int wm   = wid >> 1;        // 0..3  -> 32 rows each
  const int wn   = wid & 1;         // 0..1  -> 64 cols each
  const int rsel = lane & 15;
  const int kg   = lane >> 4;

  const float* wge = wg + (size_t)e * DD * FF;
  const float* wue = wu + (size_t)e * DD * FF;

  // Gather x row pointers once (rows fixed across K loop).
  const float* aptr[2];
#pragma unroll
  for (int i = 0; i < 2; ++i) {
    int c = i * 256 + tid;          // 512 chunks of 8 elems
    int tok = token_list[r0 + (c >> 2)];
    if (tok < 0) tok = 0;           // padded row: load row 0, never stored
    aptr[i] = x + (size_t)tok * DD + (c & 3) * 8;
  }

  v8f cg[2][4], cu[2][4];
#pragma unroll
  for (int i = 0; i < 2; ++i)
#pragma unroll
    for (int j = 0; j < 4; ++j) { cg[i][j] = (v8f)0.0f; cu[i][j] = (v8f)0.0f; }

  for (int k0 = 0; k0 < DD; k0 += TK) {
    __syncthreads();
    // --- A tile: gathered x rows, fp32 -> bf16, row-major [TM][TK] ---------
#pragma unroll
    for (int i = 0; i < 2; ++i) {
      int c = i * 256 + tid;
      const float* p = aptr[i] + k0;
      f32x4 a0 = ((const f32x4*)p)[0];
      f32x4 a1 = ((const f32x4*)p)[1];
      if (k0 + TK < DD) __builtin_prefetch(p + TK, 0, 0);
      Pack8 pk;
#pragma unroll
      for (int j = 0; j < 4; ++j) { pk.h[j] = (bf16)a0[j]; pk.h[4 + j] = (bf16)a1[j]; }
      *(u32x4*)(lA + (c >> 2) * TK + (c & 3) * 8) = pk.q;
    }
    // --- B tiles: Wg/Wu, transpose into K-contiguous [TN][TK] --------------
#pragma unroll
    for (int i = 0; i < 2; ++i) {
      int c  = i * 256 + tid;       // 512 chunks
      int kk = c >> 4;              // 0..31
      int n8 = (c & 15) * 8;        // 0..120
      const float* pg = wge + (size_t)(k0 + kk) * FF + f0 + n8;
      const float* pu = wue + (size_t)(k0 + kk) * FF + f0 + n8;
      f32x4 g0 = ((const f32x4*)pg)[0], g1 = ((const f32x4*)pg)[1];
      f32x4 u0 = ((const f32x4*)pu)[0], u1 = ((const f32x4*)pu)[1];
      if (k0 + TK < DD) {
        __builtin_prefetch(pg + (size_t)TK * FF, 0, 0);
        __builtin_prefetch(pu + (size_t)TK * FF, 0, 0);
      }
#pragma unroll
      for (int j = 0; j < 4; ++j) {
        lG[(n8 + j) * TK + kk]     = (bf16)g0[j];
        lG[(n8 + 4 + j) * TK + kk] = (bf16)g1[j];
        lU[(n8 + j) * TK + kk]     = (bf16)u0[j];
        lU[(n8 + 4 + j) * TK + kk] = (bf16)u1[j];
      }
    }
    __syncthreads();
    // --- WMMA: dual accumulators share the A fragment ----------------------
    v16bf af[2];
#pragma unroll
    for (int i = 0; i < 2; ++i) af[i] = load_frag(lA, wm * 32 + i * 16 + rsel, kg);
#pragma unroll
    for (int j = 0; j < 4; ++j) {
      int n = wn * 64 + j * 16 + rsel;
      v16bf bgv = load_frag(lG, n, kg);
      v16bf buv = load_frag(lU, n, kg);
#pragma unroll
      for (int i = 0; i < 2; ++i) {
        cg[i][j] = __builtin_amdgcn_wmma_f32_16x16x32_bf16(
            false, af[i], false, bgv, (short)0, cg[i][j], false, false);
        cu[i][j] = __builtin_amdgcn_wmma_f32_16x16x32_bf16(
            false, af[i], false, buv, (short)0, cu[i][j], false, false);
      }
    }
  }

  // --- epilogue: h = silu(g) * u, bf16 store to compacted h buffer ---------
#pragma unroll
  for (int i = 0; i < 2; ++i)
#pragma unroll
    for (int j = 0; j < 4; ++j)
#pragma unroll
      for (int v = 0; v < 8; ++v) {
        float g = cg[i][j][v];
        float u = cu[i][j][v];
        float h = (g / (1.0f + expf(-g))) * u;
        int r = r0 + wm * 32 + i * 16 + kg * 8 + v;   // C layout: M = v + 8*kg
        int f = f0 + wn * 64 + j * 16 + rsel;         //           N = lane%16
        hbuf[(size_t)r * FF + f] = (bf16)h;
      }
}

// ---------------- FFN2: out[token] = h @ Wdown ------------------------------

__global__ __launch_bounds__(256)
void moe_ffn2(const bf16* __restrict__ hbuf, const float* __restrict__ wd,
              const int* __restrict__ token_list, const int* __restrict__ offP,
              float* __restrict__ out) {
  __shared__ __align__(16) bf16 lA[TM * TK];
  __shared__ __align__(16) bf16 lB[TN * TK];

  const int r0 = blockIdx.x * TM;
  const int d0 = blockIdx.y * TN;
  const int e = find_expert(offP, r0);
  if (e < 0) return;

  const int tid  = threadIdx.x;
  const int lane = tid & 31;
  const int wid  = tid >> 5;
  const int wm   = wid >> 1;
  const int wn   = wid & 1;
  const int rsel = lane & 15;
  const int kg   = lane >> 4;

  const float* wde = wd + (size_t)e * FF * DD;

  v8f acc[2][4];
#pragma unroll
  for (int i = 0; i < 2; ++i)
#pragma unroll
    for (int j = 0; j < 4; ++j) acc[i][j] = (v8f)0.0f;

  for (int k0 = 0; k0 < FF; k0 += TK) {
    __syncthreads();
    // --- A tile: h rows already bf16 -> async DMA straight into LDS --------
#pragma unroll
    for (int i = 0; i < 2; ++i) {
      int c = i * 256 + tid;
      const bf16* src = hbuf + (size_t)(r0 + (c >> 2)) * FF + k0 + (c & 3) * 8;
      async_copy_b128(src, lA + (c >> 2) * TK + (c & 3) * 8);
    }
    // --- B tile: Wdown [F][D] fp32, transpose to [TN=d][TK=f] --------------
#pragma unroll
    for (int i = 0; i < 2; ++i) {
      int c  = i * 256 + tid;
      int kk = c >> 4;
      int n8 = (c & 15) * 8;
      const float* pb = wde + (size_t)(k0 + kk) * DD + d0 + n8;
      f32x4 b0 = ((const f32x4*)pb)[0], b1 = ((const f32x4*)pb)[1];
      if (k0 + TK < FF) __builtin_prefetch(pb + (size_t)TK * DD, 0, 0);
#pragma unroll
      for (int j = 0; j < 4; ++j) {
        lB[(n8 + j) * TK + kk]     = (bf16)b0[j];
        lB[(n8 + 4 + j) * TK + kk] = (bf16)b1[j];
      }
    }
    async_wait_all();   // ASYNCcnt -> 0: our lane's DMA chunks landed in LDS
    __syncthreads();    // all waves' chunks visible
    v16bf af[2];
#pragma unroll
    for (int i = 0; i < 2; ++i) af[i] = load_frag(lA, wm * 32 + i * 16 + rsel, kg);
#pragma unroll
    for (int j = 0; j < 4; ++j) {
      v16bf bv = load_frag(lB, wn * 64 + j * 16 + rsel, kg);
#pragma unroll
      for (int i = 0; i < 2; ++i)
        acc[i][j] = __builtin_amdgcn_wmma_f32_16x16x32_bf16(
            false, af[i], false, bv, (short)0, acc[i][j], false, false);
    }
  }

  // --- epilogue: scatter fp32 to out[token], skip padded rows --------------
#pragma unroll
  for (int i = 0; i < 2; ++i)
#pragma unroll
    for (int j = 0; j < 4; ++j)
#pragma unroll
      for (int v = 0; v < 8; ++v) {
        int r = r0 + wm * 32 + i * 16 + kg * 8 + v;
        int tok = token_list[r];
        if (tok >= 0) {
          int d = d0 + wn * 64 + j * 16 + rsel;
          out[(size_t)tok * DD + d] = acc[i][j][v];
        }
      }
}

// ---------------- launch -----------------------------------------------------

extern "C" void kernel_launch(void* const* d_in, const int* in_sizes, int n_in,
                              void* d_out, int out_size, void* d_ws, size_t ws_size,
                              hipStream_t stream) {
  const float* x   = (const float*)d_in[0];
  const int*   idx = (const int*)d_in[1];   // expert_idx (harness: int32)
  const float* wg  = (const float*)d_in[2];
  const float* wu  = (const float*)d_in[3];
  const float* wd  = (const float*)d_in[4];
  float* out = (float*)d_out;

  const int T    = in_sizes[1];             // B*S tokens
  const int Tpad = T + NE * TM;             // worst-case per-expert padding

  // Workspace layout
  char* ws = (char*)d_ws;
  int* token_list = (int*)ws;
  size_t o = ((size_t)Tpad * sizeof(int) + 255) & ~(size_t)255;
  int* counts = (int*)(ws + o); o += 256;
  int* cursor = (int*)(ws + o); o += 256;
  int* offP   = (int*)(ws + o); o += 256;
  bf16* hbuf  = (bf16*)(ws + o);            // Tpad x FF bf16 (~75 MB)

  moe_init<<<(Tpad + 255) / 256, 256, 0, stream>>>(token_list, counts, Tpad);
  moe_count<<<(T + 255) / 256, 256, 0, stream>>>(idx, counts, T);
  moe_scan<<<1, 32, 0, stream>>>(counts, offP, cursor);
  moe_scatter<<<(T + 255) / 256, 256, 0, stream>>>(idx, cursor, token_list, T);

  dim3 g1(Tpad / TM, FF / TN);   // 72 x 32
  moe_ffn1<<<g1, 256, 0, stream>>>(x, wg, wu, token_list, offP, hbuf);

  dim3 g2(Tpad / TM, DD / TN);   // 72 x 8
  moe_ffn2<<<g2, 256, 0, stream>>>(hbuf, wd, token_list, offP, out);
}
